// DotAttentionLayer_18021682774188
// MI455X (gfx1250) — matrix operand order repaired and checked by reading, not compiled, not run on hardware
//
#include <hip/hip_runtime.h>

// Problem constants (match reference)
#define BATCH 8
#define NUM_HEAD 16
#define SEQ 1024
#define DIN 1024
#define EDIM 64

typedef unsigned short u16;
typedef __attribute__((ext_vector_type(8)))  unsigned short u16x8;
typedef __attribute__((ext_vector_type(16))) __bf16        v16bf;
typedef __attribute__((ext_vector_type(8)))  float         v8f;
typedef int v4i __attribute__((__vector_size__(16)));

#if defined(__has_builtin)
#if __has_builtin(__builtin_amdgcn_global_load_async_to_lds_b128)
#define HAVE_ASYNC_LDS 1
#endif
#endif

union Frag {
  v16bf bf;
  u16x8 u[2];
};

__device__ __forceinline__ u16 f2bf(float f) {
  union { float f; unsigned int u; } v; v.f = f;
  unsigned int u = v.u;
  u += 0x7FFFu + ((u >> 16) & 1u);   // round-to-nearest-even
  return (u16)(u >> 16);
}

__device__ __forceinline__ v8f vzero8() {
  v8f z;
#pragma unroll
  for (int i = 0; i < 8; ++i) z[i] = 0.0f;
  return z;
}

__device__ __forceinline__ v8f wmma_bf16(v16bf a, v16bf b, v8f c) {
  // (neg_a, A, neg_b, B, c_mod, C, reuse_a, reuse_b)
  return __builtin_amdgcn_wmma_f32_16x16x32_bf16(false, a, false, b, (short)0, c,
                                                 false, false);
}

// 16-byte global -> LDS copy; async (ASYNCcnt-tracked) when available.
__device__ __forceinline__ void copy16_g2l(const u16* g, u16* l) {
#ifdef HAVE_ASYNC_LDS
  __builtin_amdgcn_global_load_async_to_lds_b128(
      (__attribute__((address_space(1))) v4i*)g,
      (__attribute__((address_space(3))) v4i*)l, 0, 0);
#else
  *(u16x8*)l = *(const u16x8*)g;
#endif
}

__device__ __forceinline__ void wait_async0() {
#ifdef HAVE_ASYNC_LDS
#if __has_builtin(__builtin_amdgcn_s_wait_asynccnt)
  __builtin_amdgcn_s_wait_asynccnt(0);
#else
  asm volatile("s_wait_asynccnt 0" ::: "memory");
#endif
#endif
}

// ---------------------------------------------------------------------------
// Kernel 1: f32 -> bf16 cast, 8 elements/thread
// ---------------------------------------------------------------------------
__global__ __launch_bounds__(256) void cast_f32_bf16(const float* __restrict__ src,
                                                     u16* __restrict__ dst, int n) {
  int i = (blockIdx.x * blockDim.x + threadIdx.x) * 8;
  if (i >= n) return;
  float4 a = *(const float4*)(src + i);
  float4 b = *(const float4*)(src + i + 4);
  u16x8 o;
  o[0] = f2bf(a.x); o[1] = f2bf(a.y); o[2] = f2bf(a.z); o[3] = f2bf(a.w);
  o[4] = f2bf(b.x); o[5] = f2bf(b.y); o[6] = f2bf(b.z); o[7] = f2bf(b.w);
  *(u16x8*)(dst + i) = o;
}

// ---------------------------------------------------------------------------
// Kernel 2: fused QKV projection. W chunks ([3][64][64] bf16 = 24 KB) are
// double-buffered in LDS via async global->LDS copies and shared by all 8
// waves; A fragments (per-wave X rows) come straight from global.
//   Q[b][h][s][e] row-major;  KT[b][h][e][s] (transposed);  V[b][h][s][e].
// Grid: (SEQ/128, NUM_HEAD, BATCH), 256 threads (8 waves x 16 rows).
// ---------------------------------------------------------------------------
__global__ __launch_bounds__(256) void qkv_proj(const u16* __restrict__ xb,
                                                const u16* __restrict__ wq,
                                                const u16* __restrict__ wk,
                                                const u16* __restrict__ wv,
                                                u16* __restrict__ Qo,
                                                u16* __restrict__ KTo,
                                                u16* __restrict__ Vo) {
  __shared__ __align__(16) u16 Wlds[2][3][64][72];  // 72-pitch vs bank conflicts

  const int tid  = threadIdx.x;
  const int w    = tid >> 5;
  const int lane = tid & 31;
  const int g    = lane >> 4;
  const int ln   = lane & 15;
  const int h    = blockIdx.y;
  const int b    = blockIdx.z;
  const int m0   = blockIdx.x * 128 + w * 16;

  const u16* xrow = xb + ((size_t)b * SEQ + (size_t)(m0 + ln)) * DIN;
  const u16* wbase[3];
  wbase[0] = wq + (size_t)h * DIN * EDIM;
  wbase[1] = wk + (size_t)h * DIN * EDIM;
  wbase[2] = wv + (size_t)h * DIN * EDIM;

  // Stage one 64-deep K-chunk of all three weights: 3*64 rows * 8 segs of 16B
  // = 1536 segments; 6 per thread.
  auto stageW = [&](int buf, int k0) {
#pragma unroll
    for (int i = 0; i < 6; ++i) {
      int sidx = tid + 256 * i;
      int p    = sidx >> 9;
      int rem  = sidx & 511;
      int row  = rem >> 3;
      int seg  = rem & 7;
      copy16_g2l(wbase[p] + (size_t)(k0 + row) * EDIM + seg * 8,
                 &Wlds[buf][p][row][seg * 8]);
    }
  };

  v8f acc[3][4];
#pragma unroll
  for (int p = 0; p < 3; ++p)
#pragma unroll
    for (int j = 0; j < 4; ++j) acc[p][j] = vzero8();

  stageW(0, 0);
  wait_async0();
  __syncthreads();

  int buf = 0;
  for (int c = 0; c < DIN / 64; ++c) {
    const int k0 = c * 64;
    if (c + 1 < DIN / 64) stageW(buf ^ 1, k0 + 64);   // prefetch next chunk

#pragma unroll
    for (int t = 0; t < 2; ++t) {
      Frag a;  // A-frag: lane%16 = M row; K runs [8g,+8) and [16+8g,+8)
      a.u[0] = *(const u16x8*)(xrow + k0 + 32 * t + 8 * g);
      a.u[1] = *(const u16x8*)(xrow + k0 + 32 * t + 16 + 8 * g);
#pragma unroll
      for (int p = 0; p < 3; ++p) {
        Frag bfr[4];
#pragma unroll
        for (int j = 0; j < 4; ++j) {  // B-frag: lane%16 = N col; K runs [8g,+8) and [16+8g,+8)
#pragma unroll
          for (int s = 0; s < 16; ++s)
            bfr[j].u[s >> 3][s & 7] = Wlds[buf][p][32 * t + 8 * g + (s & 7) + 16 * (s >> 3)][16 * j + ln];
        }
#pragma unroll
        for (int j = 0; j < 4; ++j)
          acc[p][j] = wmma_bf16(a.bf, bfr[j].bf, acc[p][j]);
      }
    }

    wait_async0();
    __syncthreads();
    buf ^= 1;
  }

  const size_t bh = (size_t)b * NUM_HEAD + h;
  u16* qbase = Qo + bh * SEQ * EDIM;
  u16* vbase = Vo + bh * SEQ * EDIM;
#pragma unroll
  for (int j = 0; j < 4; ++j)
#pragma unroll
    for (int r = 0; r < 8; ++r) {
      size_t off = (size_t)(m0 + r + 8 * g) * EDIM + 16 * j + ln;
      qbase[off] = f2bf(acc[0][j][r]);
      vbase[off] = f2bf(acc[2][j][r]);
    }
  u16* ktbase = KTo + bh * EDIM * SEQ;
#pragma unroll
  for (int j = 0; j < 4; ++j) {
    u16x8 pk;
#pragma unroll
    for (int r = 0; r < 8; ++r) pk[r] = f2bf(acc[1][j][r]);
    *(u16x8*)(ktbase + (size_t)(16 * j + ln) * SEQ + m0 + 8 * g) = pk;
  }
}

// ---------------------------------------------------------------------------
// Kernel 3: flash attention. KT/V 64-column tiles (8 KB each) double-buffered
// in LDS via async copies, shared by all 8 waves; online softmax per 16-row
// wave tile; P re-laid out through per-wave LDS for the PV WMMA.
// Grid: (SEQ/128, NUM_HEAD, BATCH), 256 threads.
// ---------------------------------------------------------------------------
__global__ __launch_bounds__(256) void flash_attn(const u16* __restrict__ Qb,
                                                  const u16* __restrict__ KTb,
                                                  const u16* __restrict__ Vb,
                                                  float* __restrict__ out) {
  __shared__ __align__(16) u16 Klds[2][64][72];
  __shared__ __align__(16) u16 Vlds[2][64][72];
  __shared__ __align__(16) u16 Plds[8][16][72];

  const int tid  = threadIdx.x;
  const int w    = tid >> 5;
  const int lane = tid & 31;
  const int g    = lane >> 4;
  const int ln   = lane & 15;
  const int h    = blockIdx.y;
  const int b    = blockIdx.z;
  const int m0   = blockIdx.x * 128 + w * 16;
  const size_t bh = (size_t)b * NUM_HEAD + h;

  const u16* q  = Qb + bh * SEQ * EDIM;
  const u16* kt = KTb + bh * EDIM * SEQ;
  const u16* v  = Vb + bh * SEQ * EDIM;

  // Stage KT[0:64][n0:n0+64] and V[n0:n0+64][0:64]: 512 16B segments each,
  // 2 + 2 per thread.
  auto stageKV = [&](int buf, int n0) {
#pragma unroll
    for (int i = 0; i < 2; ++i) {
      int sidx = tid + 256 * i;     // 0..511
      int row  = sidx >> 3;
      int seg  = sidx & 7;
      copy16_g2l(kt + (size_t)row * SEQ + n0 + seg * 8, &Klds[buf][row][seg * 8]);
      copy16_g2l(v + (size_t)(n0 + row) * EDIM + seg * 8, &Vlds[buf][row][seg * 8]);
    }
  };

  // Q A-fragments for the two K=32 steps over E=64 (live for the whole loop)
  Frag qf[2];
  const u16* qrow = q + (size_t)(m0 + ln) * EDIM;
#pragma unroll
  for (int t = 0; t < 2; ++t) {
    qf[t].u[0] = *(const u16x8*)(qrow + 32 * t + 8 * g);
    qf[t].u[1] = *(const u16x8*)(qrow + 32 * t + 16 + 8 * g);
  }

  v8f oacc[4];
#pragma unroll
  for (int j = 0; j < 4; ++j) oacc[j] = vzero8();
  float mrow[8], lrow[8];
#pragma unroll
  for (int r = 0; r < 8; ++r) { mrow[r] = -1.0e30f; lrow[r] = 0.0f; }

  const float scale = 0.125f;  // 1/sqrt(64)

  stageKV(0, 0);
  wait_async0();
  __syncthreads();

  int buf = 0;
  for (int n0 = 0; n0 < SEQ; n0 += 64) {
    if (n0 + 64 < SEQ) stageKV(buf ^ 1, n0 + 64);   // prefetch next KV tile

    // ---- S = (Q . K^T) * scale : 4 N-tiles x 2 K-steps = 8 WMMAs ----
    v8f s[4];
#pragma unroll
    for (int j = 0; j < 4; ++j) s[j] = vzero8();
#pragma unroll
    for (int t = 0; t < 2; ++t) {
      Frag kb[4];
#pragma unroll
      for (int j = 0; j < 4; ++j) {
#pragma unroll
        for (int s = 0; s < 16; ++s)
          kb[j].u[s >> 3][s & 7] = Klds[buf][32 * t + 8 * g + (s & 7) + 16 * (s >> 3)][16 * j + ln];
      }
#pragma unroll
      for (int j = 0; j < 4; ++j) s[j] = wmma_bf16(qf[t].bf, kb[j].bf, s[j]);
    }
#pragma unroll
    for (int j = 0; j < 4; ++j)
#pragma unroll
      for (int r = 0; r < 8; ++r) s[j][r] *= scale;

    // ---- online softmax: each row lives across 16 lanes of a half-wave ----
    float mnew[8], alpha[8];
#pragma unroll
    for (int r = 0; r < 8; ++r) {
      float tm = fmaxf(fmaxf(s[0][r], s[1][r]), fmaxf(s[2][r], s[3][r]));
      tm = fmaxf(tm, __shfl_xor(tm, 1, 32));
      tm = fmaxf(tm, __shfl_xor(tm, 2, 32));
      tm = fmaxf(tm, __shfl_xor(tm, 4, 32));
      tm = fmaxf(tm, __shfl_xor(tm, 8, 32));
      mnew[r]  = fmaxf(mrow[r], tm);
      alpha[r] = __expf(mrow[r] - mnew[r]);
      mrow[r]  = mnew[r];
    }
#pragma unroll
    for (int j = 0; j < 4; ++j)
#pragma unroll
      for (int r = 0; r < 8; ++r) s[j][r] = __expf(s[j][r] - mnew[r]);
#pragma unroll
    for (int r = 0; r < 8; ++r) {
      float ts = s[0][r] + s[1][r] + s[2][r] + s[3][r];
      ts += __shfl_xor(ts, 1, 32);
      ts += __shfl_xor(ts, 2, 32);
      ts += __shfl_xor(ts, 4, 32);
      ts += __shfl_xor(ts, 8, 32);
      lrow[r] = lrow[r] * alpha[r] + ts;
#pragma unroll
      for (int j = 0; j < 4; ++j) oacc[j][r] *= alpha[r];
    }

    // ---- re-layout P (C-format) -> A-fragment format via per-wave LDS ----
#pragma unroll
    for (int j = 0; j < 4; ++j)
#pragma unroll
      for (int r = 0; r < 8; ++r)
        Plds[w][r + 8 * g][16 * j + ln] = f2bf(s[j][r]);
    asm volatile("s_wait_dscnt 0" ::: "memory");  // same-wave LDS RAW

    // ---- O += P . V : 2 K-steps x 4 E-tiles = 8 WMMAs ----
#pragma unroll
    for (int t = 0; t < 2; ++t) {
      Frag pa;
      pa.u[0] = *(const u16x8*)&Plds[w][ln][32 * t + 8 * g];
      pa.u[1] = *(const u16x8*)&Plds[w][ln][32 * t + 16 + 8 * g];
      Frag vbf[4];
#pragma unroll
      for (int j = 0; j < 4; ++j) {
#pragma unroll
        for (int s = 0; s < 16; ++s)
          vbf[j].u[s >> 3][s & 7] = Vlds[buf][32 * t + 8 * g + (s & 7) + 16 * (s >> 3)][16 * j + ln];
      }
#pragma unroll
      for (int j = 0; j < 4; ++j) oacc[j] = wmma_bf16(pa.bf, vbf[j].bf, oacc[j]);
    }

    wait_async0();
    __syncthreads();
    buf ^= 1;
  }

  // ---- epilogue: O / l, f32 output in [B][H][S][E] order ----
  float* obase = out + bh * SEQ * EDIM;
#pragma unroll
  for (int j = 0; j < 4; ++j)
#pragma unroll
    for (int r = 0; r < 8; ++r)
      obase[(size_t)(m0 + r + 8 * g) * EDIM + 16 * j + ln] =
          oacc[j][r] / lrow[r];
}

// ---------------------------------------------------------------------------
// Host launcher
// ---------------------------------------------------------------------------
extern "C" void kernel_launch(void* const* d_in, const int* in_sizes, int n_in,
                              void* d_out, int out_size, void* d_ws, size_t ws_size,
                              hipStream_t stream) {
  const float* x  = (const float*)d_in[0];  // [B][S][D]
  const float* WQ = (const float*)d_in[1];  // [H][D][E]
  const float* WK = (const float*)d_in[2];
  const float* WV = (const float*)d_in[3];
  float* out = (float*)d_out;               // [B][H*S*E]

  // Workspace layout (u16 elements): x_bf | wq_bf | wk_bf | wv_bf | Q | KT | V
  const size_t XN = (size_t)BATCH * SEQ * DIN;              // 8M
  const size_t WN = (size_t)NUM_HEAD * DIN * EDIM;          // 1M each
  const size_t QN = (size_t)BATCH * NUM_HEAD * SEQ * EDIM;  // 8M each
  u16* xb  = (u16*)d_ws;
  u16* wqb = xb + XN;
  u16* wkb = wqb + WN;
  u16* wvb = wkb + WN;
  u16* Qb  = wvb + WN;
  u16* KTb = Qb + QN;
  u16* Vb  = KTb + QN;
  (void)ws_size; (void)in_sizes; (void)n_in; (void)out_size;  // ~74 MB used

  cast_f32_bf16<<<(int)(XN / 8 / 256), 256, 0, stream>>>(x, xb, (int)XN);
  cast_f32_bf16<<<(int)(WN / 8 / 256), 256, 0, stream>>>(WQ, wqb, (int)WN);
  cast_f32_bf16<<<(int)(WN / 8 / 256), 256, 0, stream>>>(WK, wkb, (int)WN);
  cast_f32_bf16<<<(int)(WN / 8 / 256), 256, 0, stream>>>(WV, wvb, (int)WN);

  dim3 grid(SEQ / 128, NUM_HEAD, BATCH);
  qkv_proj<<<grid, 256, 0, stream>>>(xb, wqb, wkb, wvb, Qb, KTb, Vb);
  flash_attn<<<grid, 256, 0, stream>>>(Qb, KTb, Vb, out);
}
